// MolGNN_56203942035961
// MI455X (gfx1250) — matrix operand order repaired and checked by reading, not compiled, not run on hardware
//
#include <hip/hip_runtime.h>
#include <hip/hip_bf16.h>
#include <stdint.h>

#define N_NODES 100000
#define N_EDGES 1600000
#define F_IN    11
#define HID     128
#define N_OUT   12
#define N_GRAPH 4096
#define N_LAYERS 12
#define LN_EPS  1e-5f

typedef __attribute__((ext_vector_type(16))) __bf16 v16bf;
typedef __attribute__((ext_vector_type(8)))  __bf16 v8bf;
typedef __attribute__((ext_vector_type(8)))  float  v8f;

static __device__ inline uint16_t f2bf_u16(float f) {
    union { float f; uint32_t u; } x; x.f = f;
    return (uint16_t)((x.u + 0x7FFFu + ((x.u >> 16) & 1u)) >> 16);  // round-nearest-even
}

static __device__ inline float bf2f(uint32_t lo16) {
    return __uint_as_float(lo16 << 16);
}

// ---------------------------------------------------------------- zero
__global__ void zero_f4_kernel(float4* __restrict__ p, int n4) {
    int i = blockIdx.x * blockDim.x + threadIdx.x;
    if (i < n4) p[i] = make_float4(0.f, 0.f, 0.f, 0.f);
}

// ---------------------------------------------------------------- layernorm (one wave32 per node) -> bf16 h
__global__ void layernorm_kernel(const float* __restrict__ src, int ld, int F,
                                 const float* __restrict__ gamma,
                                 const float* __restrict__ beta,
                                 uint16_t* __restrict__ h) {
    int wave = threadIdx.x >> 5;
    int lane = threadIdx.x & 31;
    int node = blockIdx.x * 8 + wave;
    if (node >= N_NODES) return;
    const float* row = src + (size_t)node * ld;
    float v[4];
    float sum = 0.f;
#pragma unroll
    for (int i = 0; i < 4; ++i) {
        int c = lane + i * 32;
        v[i] = (c < F) ? row[c] : 0.f;
        sum += v[i];
    }
#pragma unroll
    for (int off = 16; off > 0; off >>= 1) sum += __shfl_xor(sum, off, 32);
    float mu = sum / (float)F;
    float vs = 0.f;
#pragma unroll
    for (int i = 0; i < 4; ++i) {
        int c = lane + i * 32;
        float d = (c < F) ? (v[i] - mu) : 0.f;
        vs += d * d;
    }
#pragma unroll
    for (int off = 16; off > 0; off >>= 1) vs += __shfl_xor(vs, off, 32);
    float rs = rsqrtf(vs / (float)F + LN_EPS);
    uint16_t* hr = h + (size_t)node * HID;
#pragma unroll
    for (int i = 0; i < 4; ++i) {
        int c = lane + i * 32;
        hr[c] = (c < F) ? f2bf_u16((v[i] - mu) * rs * gamma[c] + beta[c]) : (uint16_t)0;
    }
}

// ---------------------------------------------------------------- edge aggregation (bf16 gather, f32 atomics, L2-resident)
__global__ void agg128_kernel(const int* __restrict__ src, const int* __restrict__ dst,
                              const uint16_t* __restrict__ hbf, float* __restrict__ agg) {
    long long t = (long long)blockIdx.x * blockDim.x + threadIdx.x;
    if (t >= (long long)N_EDGES * 32) return;
    int e = (int)(t >> 5);
    int q = (int)(t & 31);
    int s = src[e], d = dst[e];
    const uint32_t* hp = (const uint32_t*)(hbf + (size_t)s * HID + q * 4);  // 4 bf16 = 8 B
    uint32_t w0 = hp[0], w1 = hp[1];
    float* ap = agg + (size_t)d * HID + q * 4;
    atomicAdd(ap + 0, bf2f(w0 & 0xFFFFu));
    atomicAdd(ap + 1, __uint_as_float(w0 & 0xFFFF0000u));
    atomicAdd(ap + 2, bf2f(w1 & 0xFFFFu));
    atomicAdd(ap + 3, __uint_as_float(w1 & 0xFFFF0000u));
}

__global__ void agg_small_kernel(const int* __restrict__ src, const int* __restrict__ dst,
                                 const uint16_t* __restrict__ hbf, float* __restrict__ agg, int F) {
    int e = blockIdx.x * blockDim.x + threadIdx.x;
    if (e >= N_EDGES) return;
    int s = src[e], d = dst[e];
    const uint16_t* hr = hbf + (size_t)s * HID;
    float* ar = agg + (size_t)d * HID;
    for (int c = 0; c < F; ++c) atomicAdd(ar + c, bf2f((uint32_t)hr[c]));
}

// ---------------------------------------------------------------- f32 -> bf16 pass for the aggregated features
__global__ void cvt_bf16_kernel(const float* __restrict__ in, uint16_t* __restrict__ outp, int n4) {
    int i = blockIdx.x * blockDim.x + threadIdx.x;
    if (i >= n4) return;
    float4 v = ((const float4*)in)[i];
    uint32_t p0 = ((uint32_t)f2bf_u16(v.y) << 16) | f2bf_u16(v.x);
    uint32_t p1 = ((uint32_t)f2bf_u16(v.w) << 16) | f2bf_u16(v.z);
    ((uint2*)outp)[i] = make_uint2(p0, p1);
}

// ---------------------------------------------------------------- weight pre-convert + B-fragment swizzle
// Combined weights per layer: rows [0,Khalf) = W_rel (zero-padded past Kin), rows [Khalf,2*Khalf) = W_root.
// Stored so each lane's 16-element B fragment (s,nt,lane) is contiguous:
//   value = Wcomb[s*32 + (lane/16)*16 + j][nt*16 + lane%16]
__global__ void wconv_kernel(const float* __restrict__ Wrel0, const float* __restrict__ Wroot0,
                             const float* __restrict__ Wrel,  const float* __restrict__ Wroot,
                             uint16_t* __restrict__ wbf) {
    int layer  = blockIdx.y;
    int Ksteps = (layer == 0) ? 2 : 8;
    int Khalf  = (layer == 0) ? 32 : 128;
    int Kin    = (layer == 0) ? F_IN : HID;
    int gid = blockIdx.x * blockDim.x + threadIdx.x;
    if (gid >= Ksteps * 256) return;
    int s    = gid >> 8;
    int rem  = gid & 255;
    int nt   = rem >> 5;
    int lane = rem & 31;
    const float* Wr = (layer == 0) ? Wrel0  : Wrel  + (size_t)(layer - 1) * HID * HID;
    const float* Wt = (layer == 0) ? Wroot0 : Wroot + (size_t)(layer - 1) * HID * HID;
    int col = nt * 16 + (lane & 15);
    int kb  = s * 32 + (lane >> 4) * 16;
    uint16_t* out = wbf + (size_t)layer * 32768 + ((size_t)(s * 8 + nt) * 32 + lane) * 16;
    for (int j = 0; j < 16; ++j) {
        int k = kb + j;
        float val;
        if (k < Khalf) {
            val = (k < Kin) ? Wr[(size_t)k * HID + col] : 0.f;
        } else {
            int k2 = k - Khalf;
            val = (k2 < Kin) ? Wt[(size_t)k2 * HID + col] : 0.f;
        }
        out[j] = f2bf_u16(val);
    }
}

// ---------------------------------------------------------------- fused dual-GEMM + bias + ReLU + residual
// x = relu([agg|h] @ [W_rel;W_root] + b)   (+ residual handling via `mode`)
// block = 256 thr (8 waves); block tile = 128 rows x 128 cols; wave tile = 16 x 128.
__global__ __launch_bounds__(256) void gemm_layer_kernel(
        const uint16_t* __restrict__ aggbf, const uint16_t* __restrict__ hbf,
        const uint16_t* __restrict__ wbf, const float* __restrict__ bias,
        float* __restrict__ x, float* __restrict__ res,
        int Ksteps, int mode) {
    __shared__ __align__(32) uint16_t sB[32768];   // 64 KB swizzled bf16 weights
    int tid  = threadIdx.x;
    int wave = tid >> 5;
    int lane = tid & 31;

    // cooperative LDS stage of pre-swizzled bf16 weights
    {
        const int4* gw = (const int4*)wbf;
        int4* sw = (int4*)sB;
        int n4 = Ksteps * 512;
        for (int i = tid; i < n4; i += 256) sw[i] = gw[i];
    }
    __syncthreads();

    int KhSteps = Ksteps >> 1;
    int rowA = blockIdx.x * 128 + wave * 16 + (lane & 15);
    if (rowA >= N_NODES) rowA = N_NODES - 1;          // clamp loads; stores guarded below
    const uint16_t* aggRow = aggbf + (size_t)rowA * HID;
    const uint16_t* hRow   = hbf   + (size_t)rowA * HID;
    int koff = (lane >> 4) * 16;   // A layout: lanes 0-15 -> K {0..15}; lanes 16-31 -> {16..31}

    v8f acc[8];
#pragma unroll
    for (int nt = 0; nt < 8; ++nt)
#pragma unroll
        for (int i = 0; i < 8; ++i) acc[nt][i] = 0.f;

    for (int s = 0; s < Ksteps; ++s) {
        const uint16_t* base = (s < KhSteps) ? (aggRow + s * 32) : (hRow + (s - KhSteps) * 32);
        v8bf lo = *(const v8bf*)(base + koff);          // 16 B
        v8bf hi = *(const v8bf*)(base + koff + 8);      // 16 B
        v16bf a = __builtin_shufflevector(lo, hi, 0, 1, 2, 3, 4, 5, 6, 7,
                                                  8, 9, 10, 11, 12, 13, 14, 15);
#pragma unroll
        for (int nt = 0; nt < 8; ++nt) {
            v16bf b = *(const v16bf*)(sB + ((size_t)(s * 8 + nt) * 32 + lane) * 16);
            acc[nt] = __builtin_amdgcn_wmma_f32_16x16x32_bf16(
                false, a, false, b, (short)0, acc[nt], false, false);
        }
    }

    // D layout: lane holds col = nt*16 + lane%16, rows = wave*16 + (lane/16)*8 + v
    int rowBase = blockIdx.x * 128 + wave * 16 + (lane >> 4) * 8;
    int colLane = lane & 15;
    bool fullTile = (blockIdx.x * 128 + 128) <= N_NODES;   // uniform across block

    if (fullTile) {
#pragma unroll
        for (int nt = 0; nt < 8; ++nt) {
            int col = nt * 16 + colLane;
            float bb = bias[col];
#pragma unroll
            for (int v = 0; v < 8; ++v) {
                size_t idx = (size_t)(rowBase + v) * HID + col;
                float val = fmaxf(acc[nt][v] + bb, 0.f);
                if (mode == 2) val += res[idx];        // layer 8: add residual (captured at layer 4)
                if (mode == 1) res[idx] = val;         // layer 4: capture residual
                x[idx] = val;
            }
        }
    } else {
#pragma unroll
        for (int nt = 0; nt < 8; ++nt) {
            int col = nt * 16 + colLane;
            float bb = bias[col];
#pragma unroll
            for (int v = 0; v < 8; ++v) {
                int r = rowBase + v;
                if (r < N_NODES) {
                    size_t idx = (size_t)r * HID + col;
                    float val = fmaxf(acc[nt][v] + bb, 0.f);
                    if (mode == 2) val += res[idx];
                    if (mode == 1) res[idx] = val;
                    x[idx] = val;
                }
            }
        }
    }
}

// ---------------------------------------------------------------- graph pooling + readout
__global__ void pool_kernel(const int* __restrict__ bid, const float* __restrict__ x,
                            float* __restrict__ sums) {
    long long t = (long long)blockIdx.x * blockDim.x + threadIdx.x;
    if (t >= (long long)N_NODES * 32) return;
    int n = (int)(t >> 5);
    int q = (int)(t & 31);
    int g = bid[n];
    const float4 xv = *(const float4*)(x + (size_t)n * HID + q * 4);
    float* sp = sums + (size_t)g * HID + q * 4;
    atomicAdd(sp + 0, xv.x);
    atomicAdd(sp + 1, xv.y);
    atomicAdd(sp + 2, xv.z);
    atomicAdd(sp + 3, xv.w);
}

__global__ void cnt_kernel(const int* __restrict__ bid, float* __restrict__ cnt) {
    int n = blockIdx.x * blockDim.x + threadIdx.x;
    if (n < N_NODES) atomicAdd(cnt + bid[n], 1.0f);
}

__global__ void readout_kernel(const float* __restrict__ sums, const float* __restrict__ cnt,
                               const float* __restrict__ Wout, const float* __restrict__ bout,
                               float* __restrict__ out) {
    int t = blockIdx.x * blockDim.x + threadIdx.x;
    if (t >= N_GRAPH * N_OUT) return;
    int g = t / N_OUT;
    int o = t % N_OUT;
    float inv = 1.f / fmaxf(cnt[g], 1.f);
    const float* sr = sums + (size_t)g * HID;
    float acc = 0.f;
    for (int c = 0; c < HID; ++c) acc += sr[c] * Wout[c * N_OUT + o];
    out[t] = acc * inv + bout[o];
}

// ---------------------------------------------------------------- launcher
extern "C" void kernel_launch(void* const* d_in, const int* in_sizes, int n_in,
                              void* d_out, int out_size, void* d_ws, size_t ws_size,
                              hipStream_t stream) {
    (void)in_sizes; (void)n_in; (void)out_size; (void)ws_size;
    const float* x_in   = (const float*)d_in[0];
    const int*   ei     = (const int*)d_in[1];
    const int*   srcIdx = ei;
    const int*   dstIdx = ei + N_EDGES;
    const int*   bid    = (const int*)d_in[2];
    const float* Wrel0  = (const float*)d_in[3];
    const float* Wroot0 = (const float*)d_in[4];
    const float* b0     = (const float*)d_in[5];
    const float* g0     = (const float*)d_in[6];
    const float* be0    = (const float*)d_in[7];
    const float* Wrel   = (const float*)d_in[8];
    const float* Wroot  = (const float*)d_in[9];
    const float* bvec   = (const float*)d_in[10];
    const float* gvec   = (const float*)d_in[11];
    const float* bevec  = (const float*)d_in[12];
    const float* Wout   = (const float*)d_in[13];
    const float* bout   = (const float*)d_in[14];
    float* out = (float*)d_out;

    char* ws = (char*)d_ws;
    size_t off = 0;
    const size_t NHID = (size_t)N_NODES * HID;
    float*    x_buf  = (float*)(ws + off);    off += NHID * 4;
    float*    agg    = (float*)(ws + off);    off += NHID * 4;
    float*    res    = (float*)(ws + off);    off += NHID * 4;
    uint16_t* hbf    = (uint16_t*)(ws + off); off += NHID * 2;
    uint16_t* aggbf  = (uint16_t*)(ws + off); off += NHID * 2;
    uint16_t* wbf    = (uint16_t*)(ws + off); off += (size_t)N_LAYERS * 32768 * 2;
    float*    sums   = (float*)(ws + off);    off += (size_t)N_GRAPH * HID * 4;
    float*    cnt    = (float*)(ws + off);    off += (size_t)N_GRAPH * 4;   // contiguous after sums

    // 1) convert + swizzle all layer weights to bf16 B-fragment layout
    dim3 wgrid(8, N_LAYERS);
    wconv_kernel<<<wgrid, 256, 0, stream>>>(Wrel0, Wroot0, Wrel, Wroot, wbf);

    const int nblk_rows = (N_NODES + 127) / 128;
    const int n4 = (int)(NHID / 4);
    for (int l = 0; l < N_LAYERS; ++l) {
        const float* srcx  = (l == 0) ? x_in : x_buf;
        int ld             = (l == 0) ? F_IN : HID;
        const float* gamma = (l == 0) ? g0  : gvec  + (size_t)(l - 1) * HID;
        const float* beta  = (l == 0) ? be0 : bevec + (size_t)(l - 1) * HID;
        const float* bias  = (l == 0) ? b0  : bvec  + (size_t)(l - 1) * HID;

        layernorm_kernel<<<(N_NODES + 7) / 8, 256, 0, stream>>>(srcx, ld, ld, gamma, beta, hbf);

        zero_f4_kernel<<<(n4 + 255) / 256, 256, 0, stream>>>((float4*)agg, n4);

        if (l == 0) {
            agg_small_kernel<<<(N_EDGES + 255) / 256, 256, 0, stream>>>(srcIdx, dstIdx, hbf, agg, F_IN);
        } else {
            long long tthreads = (long long)N_EDGES * 32;
            agg128_kernel<<<(int)((tthreads + 255) / 256), 256, 0, stream>>>(srcIdx, dstIdx, hbf, agg);
        }

        cvt_bf16_kernel<<<(n4 + 255) / 256, 256, 0, stream>>>(agg, aggbf, n4);

        int Ksteps = (l == 0) ? 2 : 8;
        int mode   = (l == 4) ? 1 : ((l == 8) ? 2 : 0);
        gemm_layer_kernel<<<nblk_rows, 256, 0, stream>>>(
            aggbf, hbf, wbf + (size_t)l * 32768, bias, x_buf, res, Ksteps, mode);
    }

    // 2) pooled readout
    int pz = (int)(((size_t)N_GRAPH * HID + N_GRAPH) / 4);  // zero sums + cnt together
    zero_f4_kernel<<<(pz + 255) / 256, 256, 0, stream>>>((float4*)sums, pz);
    long long pt = (long long)N_NODES * 32;
    pool_kernel<<<(int)((pt + 255) / 256), 256, 0, stream>>>(bid, x_buf, sums);
    cnt_kernel<<<(N_NODES + 255) / 256, 256, 0, stream>>>(bid, cnt);
    readout_kernel<<<(N_GRAPH * N_OUT + 255) / 256, 256, 0, stream>>>(sums, cnt, Wout, bout, out);
}
